// TRMAttention_77335181132435
// MI455X (gfx1250) — compile-verified, matrix operand order checked by reading, not run on hardware
//
#include <hip/hip_runtime.h>
#include <hip/hip_bf16.h>
#include <math.h>

#define Bv 2
#define Sv 2048
#define HIDv 1024
#define Hv 16
#define Dv 64

typedef __attribute__((ext_vector_type(16))) _Float16 v16h;
typedef __attribute__((ext_vector_type(8)))  _Float16 v8h;
typedef __attribute__((ext_vector_type(8)))  float    v8f;

#define WMMA_F16(a, b, c) \
    __builtin_amdgcn_wmma_f32_16x16x32_f16(false, (a), false, (b), (short)0, (c), false, false)

// Load one 16x32 F16 fragment (A layout; B uses the same layout with N<->M).
// rowbase must point at the start of this lane's row's 32-element K slice.
// ISA 7.12.2: lane<16 -> halves [0..7] and [16..23]; lane>=16 -> [8..15], [24..31].
__device__ __forceinline__ v16h load_frag(const _Float16* rowbase, int lane) {
    const int off = (lane < 16) ? 0 : 8;
    v8h lo = *reinterpret_cast<const v8h*>(rowbase + off);
    v8h hi = *reinterpret_cast<const v8h*>(rowbase + off + 16);
    v16h r;
#pragma unroll
    for (int i = 0; i < 8; ++i) { r[i] = lo[i]; r[i + 8] = hi[i]; }
    return r;
}

// CDNA5 async copy: 16B per lane straight into LDS, tracked by ASYNCcnt.
__device__ __forceinline__ void async_ld16(unsigned lds_byte_addr, const _Float16* gaddr) {
    asm volatile("global_load_async_to_lds_b128 %0, %1, off"
                 :: "v"(lds_byte_addr), "v"(gaddr) : "memory");
}

// ---------------- conversion / transpose kernels ----------------
__global__ void cvt_f32_to_f16(const float* __restrict__ in, _Float16* __restrict__ out, int n) {
    int i = blockIdx.x * blockDim.x + threadIdx.x;
    if (i < n) out[i] = (_Float16)in[i];
}

// in[rows][cols] (f32) -> out[cols][rows] (f16)
__global__ void transpose_to_f16(const float* __restrict__ in, _Float16* __restrict__ out,
                                 int rows, int cols) {
    int i = blockIdx.x * blockDim.x + threadIdx.x;
    if (i < rows * cols) {
        int r = i / cols, c = i % cols;
        out[(size_t)c * rows + r] = (_Float16)in[i];
    }
}

// ---------------- generic WMMA GEMM: C[M][N] = A[M][K] * Bt[N][K]^T ----------------
// 256 threads = 8 waves; each wave owns one 16x16 output tile; block covers 16(M) x 128(N).
__global__ __launch_bounds__(256) void gemm_wmma_f16(const _Float16* __restrict__ A,
                                                     const _Float16* __restrict__ Bt,
                                                     float* __restrict__ C,
                                                     int M, int N, int K) {
    const int lane  = threadIdx.x & 31;
    const int wave  = threadIdx.x >> 5;
    const int mblk  = M >> 4;
    const int mt    = blockIdx.x % mblk;
    const int nb    = blockIdx.x / mblk;
    const int m0    = mt * 16;
    const int n0    = nb * 128 + wave * 16;
    const int row15 = lane & 15;

    const _Float16* arow = A  + (size_t)(m0 + row15) * K;
    const _Float16* brow = Bt + (size_t)(n0 + row15) * K;

    v8f acc = {};
    for (int k0 = 0; k0 < K; k0 += 32) {
        v16h a = load_frag(arow + k0, lane);
        v16h b = load_frag(brow + k0, lane);
        acc = WMMA_F16(a, b, acc);
    }
    const int g = (lane < 16) ? 0 : 8;
#pragma unroll
    for (int r = 0; r < 8; ++r)
        C[(size_t)(m0 + r + g) * N + n0 + row15] = acc[r];
}

// ---------------- RoPE + layout shuffle ----------------
// QKV f32 [b*S+s][3072] -> Qh,Kh f16 [(b*H+h)*S+s][64] (rotated), Vt f16 [(b*H+h)*64+d][S]
__global__ void rope_split(const float* __restrict__ QKV,
                           const float* __restrict__ cosT, const float* __restrict__ sinT,
                           _Float16* __restrict__ Qh, _Float16* __restrict__ Kh,
                           _Float16* __restrict__ Vt) {
    int i = blockIdx.x * blockDim.x + threadIdx.x;  // b,s,h,d packed
    if (i >= Bv * Sv * Hv * Dv) return;
    const int d = i & (Dv - 1);
    const int h = (i >> 6) & (Hv - 1);
    const int s = (i >> 10) & (Sv - 1);
    const int b = i >> 21;

    const size_t rowq = (size_t)(b * Sv + s) * 3072 + h * 64;
    const int dp = (d < 32) ? d + 32 : d - 32;
    const float sgn = (d < 32) ? -1.0f : 1.0f;
    const float c  = cosT[s * 64 + d];
    const float sn = sinT[s * 64 + d];

    const float q  = QKV[rowq + d],        qp = QKV[rowq + dp];
    const float k  = QKV[rowq + 1024 + d], kp = QKV[rowq + 1024 + dp];
    const float v  = QKV[rowq + 2048 + d];

    const size_t bh  = (size_t)b * Hv + h;
    const size_t orow = (bh * Sv + s) * 64 + d;
    Qh[orow] = (_Float16)(q * c + sgn * qp * sn);
    Kh[orow] = (_Float16)(k * c + sgn * kp * sn);
    Vt[(bh * 64 + d) * Sv + s] = (_Float16)v;
}

// ---------------- flash attention ----------------
// 8 waves/block; each wave owns 16 q rows (128 q rows/block of one (b,h)).
// K/V tiles (32 keys) staged in double-buffered LDS via async-to-LDS copies.
__global__ __launch_bounds__(256) void attn_flash(const _Float16* __restrict__ Qh,
                                                  const _Float16* __restrict__ Kh,
                                                  const _Float16* __restrict__ Vt,
                                                  _Float16* __restrict__ Ah) {
    __shared__ __align__(16) _Float16 ldsK[2][32 * 64];  // [buf][key][d]   8 KB
    __shared__ __align__(16) _Float16 ldsV[2][64 * 32];  // [buf][d][key]   8 KB
    __shared__ __align__(16) _Float16 ldsP[8][16 * 32];  // per-wave probs  8 KB

    const int tid   = threadIdx.x;
    const int lane  = tid & 31;
    const int wave  = tid >> 5;
    const int nqb   = Sv / 128;                 // q-blocks per (b,h)
    const int qblk  = blockIdx.x % nqb;
    const int bh    = blockIdx.x / nqb;
    const int h     = bh & (Hv - 1);
    const int b     = bh / Hv;
    const int q0    = qblk * 128 + wave * 16;
    const int row15 = lane & 15;
    const int g     = (lane < 16) ? 0 : 8;
    const size_t bhS  = (size_t)bh * Sv;
    const size_t bh64 = (size_t)bh * 64;

    // cooperative async tile copy: 256 lanes x 16B = 4KB each for K and V
    const int krow = tid >> 3, kseg = (tid & 7) * 8;    // K: 32 rows x 64 halves
    const int vrow = tid >> 2, vseg = (tid & 3) * 8;    // V: 64 rows x 32 halves
    const _Float16* gK = Kh + (bhS + krow) * 64 + kseg;
    const _Float16* gV = Vt + (bh64 + vrow) * Sv + vseg;
    const unsigned lK = (unsigned)(uintptr_t)&ldsK[0][krow * 64 + kseg];
    const unsigned lV = (unsigned)(uintptr_t)&ldsV[0][vrow * 32 + vseg];
    const unsigned bufStrideK = (unsigned)(32 * 64 * sizeof(_Float16));
    const unsigned bufStrideV = (unsigned)(64 * 32 * sizeof(_Float16));

    const _Float16* qrow = Qh + (bhS + q0 + row15) * 64;
    v16h qf0 = load_frag(qrow, lane);
    v16h qf1 = load_frag(qrow + 32, lane);

    v8f o0 = {}, o1 = {}, o2 = {}, o3 = {};
    float m[8], l[8];
#pragma unroll
    for (int r = 0; r < 8; ++r) { m[r] = -1e30f; l[r] = 0.0f; }
    const float sc = 0.125f;  // 1/sqrt(64)

    // prime buffer 0
    async_ld16(lK, gK);
    async_ld16(lV, gV);

    for (int key0 = 0; key0 < Sv; key0 += 32) {
        const int buf = (key0 >> 5) & 1;
        if (key0 + 32 < Sv) {  // prefetch next tile into the other buffer
            async_ld16(lK + (buf ^ 1) * bufStrideK, gK + (size_t)(key0 + 32) * 64);
            async_ld16(lV + (buf ^ 1) * bufStrideV, gV + (key0 + 32));
            asm volatile("s_wait_asynccnt 0x2" ::: "memory");
        } else {
            asm volatile("s_wait_asynccnt 0x0" ::: "memory");
        }
        __syncthreads();  // tile visible to all 8 waves

        const _Float16* k0row = &ldsK[buf][row15 * 64];
        const _Float16* k1row = &ldsK[buf][(16 + row15) * 64];
        v8f s0 = {}, s1 = {};
        s0 = WMMA_F16(qf0, load_frag(k0row, lane), s0);
        s0 = WMMA_F16(qf1, load_frag(k0row + 32, lane), s0);
        s1 = WMMA_F16(qf0, load_frag(k1row, lane), s1);
        s1 = WMMA_F16(qf1, load_frag(k1row + 32, lane), s1);

#pragma unroll
        for (int r = 0; r < 8; ++r) {
            float a0 = s0[r] * sc, a1 = s1[r] * sc;
            float mx = fmaxf(a0, a1);
            mx = fmaxf(mx, __shfl_xor(mx, 1, 32));
            mx = fmaxf(mx, __shfl_xor(mx, 2, 32));
            mx = fmaxf(mx, __shfl_xor(mx, 4, 32));
            mx = fmaxf(mx, __shfl_xor(mx, 8, 32));
            float mn  = fmaxf(m[r], mx);
            float fac = expf(m[r] - mn);
            float p0  = expf(a0 - mn);
            float p1  = expf(a1 - mn);
            float rs  = p0 + p1;
            rs += __shfl_xor(rs, 1, 32);
            rs += __shfl_xor(rs, 2, 32);
            rs += __shfl_xor(rs, 4, 32);
            rs += __shfl_xor(rs, 8, 32);
            l[r] = l[r] * fac + rs;
            m[r] = mn;
            o0[r] *= fac; o1[r] *= fac; o2[r] *= fac; o3[r] *= fac;
            const int row = r + g;
            ldsP[wave][row * 32 + row15]      = (_Float16)p0;
            ldsP[wave][row * 32 + 16 + row15] = (_Float16)p1;
        }
        // wave-private LDS region; DS ops from the same wave are in-order (ISA 7.3)
        v16h pa = load_frag(&ldsP[wave][row15 * 32], lane);
        o0 = WMMA_F16(pa, load_frag(&ldsV[buf][( 0 + row15) * 32], lane), o0);
        o1 = WMMA_F16(pa, load_frag(&ldsV[buf][(16 + row15) * 32], lane), o1);
        o2 = WMMA_F16(pa, load_frag(&ldsV[buf][(32 + row15) * 32], lane), o2);
        o3 = WMMA_F16(pa, load_frag(&ldsV[buf][(48 + row15) * 32], lane), o3);
        __syncthreads();  // everyone done with this buffer before it is refilled
    }

#pragma unroll
    for (int r = 0; r < 8; ++r) {
        const float inv = 1.0f / l[r];
        const int srow = q0 + r + g;
        const size_t obase = ((size_t)b * Sv + srow) * 1024 + h * 64;
        Ah[obase +  0 + row15] = (_Float16)(o0[r] * inv);
        Ah[obase + 16 + row15] = (_Float16)(o1[r] * inv);
        Ah[obase + 32 + row15] = (_Float16)(o2[r] * inv);
        Ah[obase + 48 + row15] = (_Float16)(o3[r] * inv);
    }
}

extern "C" void kernel_launch(void* const* d_in, const int* in_sizes, int n_in,
                              void* d_out, int out_size, void* d_ws, size_t ws_size,
                              hipStream_t stream) {
    const float* hidden = (const float*)d_in[0];   // [B,S,HID]
    const float* cosT   = (const float*)d_in[1];   // [S,64]
    const float* sinT   = (const float*)d_in[2];   // [S,64]
    const float* W_qkv  = (const float*)d_in[3];   // [1024,3072]
    const float* W_o    = (const float*)d_in[4];   // [1024,1024]
    float* out = (float*)d_out;

    const size_t MB = 1u << 20;
    char* ws = (char*)d_ws;
    _Float16* Xh    = (_Float16*)(ws + 0 * MB);   //  8 MB : hidden f16 [4096][1024]
    _Float16* Wqkvt = (_Float16*)(ws + 8 * MB);   //  6 MB : W_qkv^T f16 [3072][1024]
    _Float16* Wot   = (_Float16*)(ws + 14 * MB);  //  2 MB : W_o^T  f16 [1024][1024]
    _Float16* Qh    = (_Float16*)(ws + 16 * MB);  //  8 MB : [b,h,s,d]
    _Float16* Kh    = (_Float16*)(ws + 24 * MB);  //  8 MB : [b,h,s,d]
    _Float16* Vt    = (_Float16*)(ws + 32 * MB);  //  8 MB : [b,h,d,s]
    _Float16* Ah    = (_Float16*)(ws + 40 * MB);  //  8 MB : attn out f16 [4096][1024]
    float*    QKV   = (float*)   (ws + 48 * MB);  // 48 MB : raw QKV f32 [4096][3072]

    const int M = Bv * Sv;        // 4096
    // 1) precision conversion / weight transposes
    cvt_f32_to_f16<<<(M * HIDv + 255) / 256, 256, 0, stream>>>(hidden, Xh, M * HIDv);
    transpose_to_f16<<<(HIDv * 3072 + 255) / 256, 256, 0, stream>>>(W_qkv, Wqkvt, HIDv, 3072);
    transpose_to_f16<<<(HIDv * HIDv + 255) / 256, 256, 0, stream>>>(W_o, Wot, HIDv, HIDv);

    // 2) QKV = X @ W_qkv   (M=4096, N=3072, K=1024)
    gemm_wmma_f16<<<(M / 16) * (3072 / 128), 256, 0, stream>>>(Xh, Wqkvt, QKV, M, 3072, HIDv);

    // 3) RoPE + split/transpose
    rope_split<<<(Bv * Sv * Hv * Dv + 255) / 256, 256, 0, stream>>>(QKV, cosT, sinT, Qh, Kh, Vt);

    // 4) flash attention: 8 waves/block, async double-buffered K/V tiles in LDS
    attn_flash<<<Bv * Hv * (Sv / 128), 256, 0, stream>>>(Qh, Kh, Vt, Ah);

    // 5) out = attn @ W_o  (M=4096, N=1024, K=1024)
    gemm_wmma_f16<<<(M / 16) * (HIDv / 128), 256, 0, stream>>>(Ah, Wot, out, M, HIDv, HIDv);
}
